// GNNModel_45947560132963
// MI455X (gfx1250) — matrix operand order repaired and checked by reading, not compiled, not run on hardware
//
#include <hip/hip_runtime.h>
#include <math.h>

// ---------------------------------------------------------------------------
// Graph-transformer forward (MI455X / gfx1250, wave32, WMMA bf16->f32)
//
// Shapes: N_ATOMS=4096, N_BONDS=8192, MAX_NB=12, HID=256, HEADS=8, DH=32,
//         DEPTH=3, ATOM_FD=133, BOND_FD=147.
//
// Input flattening assumption (JAX tree_leaves, dict keys sorted):
//  0 a2b[4096*12] i32   1 b2a[8192] i32   2 f_atoms[4096*133]  3 f_bonds[8192*147]
//  4 ae.lin.b  5 ae.lin.w  6 ae.ln_b  7 ae.ln_g
//  8 be.lin.b  9 be.lin.w 10 be.ln_b 11 be.ln_g
// 12 gc.a1.b 13 gc.a1.w 14 gc.a2.b 15 gc.a2.w 16 gc.t1.b 17 gc.t1.w 18 gc.t2.b 19 gc.t2.w
// 20 gf.gate.b 21 gf.gate.w
// 22+20*l per layer: ff1.b ff1.w ff2.b ff2.w k.b k.w n1b n1g n2b n2g
//                    na1.b na1.w na2.b na2.w o.b o.w q.b q.w v.b v.w
// 82 out.l1.b 83 out.l1.w 84 out.l2.b 85 out.l2.w 86 out.ln_b 87 out.ln_g
// ---------------------------------------------------------------------------

#define HID    256
#define NATOM  4096
#define NBOND  8192
#define MAXNB  12
#define HEADS  8
#define DH     32

typedef __attribute__((ext_vector_type(16))) __bf16 v16bf;
typedef __attribute__((ext_vector_type(8)))  float  v8f;

// A fragment (16x32 bf16): lane holds row (lane&15); element i maps to
// k = 16*(i>=8) + 8*(lane>=16) + (i&7)  => two contiguous 8-float runs.
__device__ __forceinline__ v16bf a_frag_vec(const float* __restrict__ p,
                                            int kk, int ksel)
{
  const float4* q0 = (const float4*)(p + kk + (ksel << 3));
  const float4* q1 = (const float4*)(p + kk + 16 + (ksel << 3));
  const float4 x0 = q0[0], x1 = q0[1], x2 = q1[0], x3 = q1[1];
  v16bf a;
  a[0]=(__bf16)x0.x;  a[1]=(__bf16)x0.y;  a[2]=(__bf16)x0.z;  a[3]=(__bf16)x0.w;
  a[4]=(__bf16)x1.x;  a[5]=(__bf16)x1.y;  a[6]=(__bf16)x1.z;  a[7]=(__bf16)x1.w;
  a[8]=(__bf16)x2.x;  a[9]=(__bf16)x2.y;  a[10]=(__bf16)x2.z; a[11]=(__bf16)x2.w;
  a[12]=(__bf16)x3.x; a[13]=(__bf16)x3.y; a[14]=(__bf16)x3.z; a[15]=(__bf16)x3.w;
  return a;
}

// B fragment (32x16 bf16): lane holds col (lane&15); element i maps to
// k = 16*(i>=8) + 8*(lane>=16) + (i&7)  => two contiguous 8-float runs.
__device__ __forceinline__ v16bf b_frag_vec(const float* __restrict__ p,
                                            int kk, int ksel)
{
  const float4* q0 = (const float4*)(p + kk + (ksel << 3));
  const float4* q1 = (const float4*)(p + kk + 16 + (ksel << 3));
  const float4 x0 = q0[0], x1 = q0[1], x2 = q1[0], x3 = q1[1];
  v16bf b;
  b[0]=(__bf16)x0.x;  b[1]=(__bf16)x0.y;  b[2]=(__bf16)x0.z;  b[3]=(__bf16)x0.w;
  b[4]=(__bf16)x1.x;  b[5]=(__bf16)x1.y;  b[6]=(__bf16)x1.z;  b[7]=(__bf16)x1.w;
  b[8]=(__bf16)x2.x;  b[9]=(__bf16)x2.y;  b[10]=(__bf16)x2.z; b[11]=(__bf16)x2.w;
  b[12]=(__bf16)x3.x; b[13]=(__bf16)x3.y; b[14]=(__bf16)x3.z; b[15]=(__bf16)x3.w;
  return b;
}

// ---------------------------------------------------------------------------
// Generic WMMA GEMM:  C[M,N] = act(alpha * A[M,K] @ W[N,K]^T + bias[N] (+ C))
//   aidx  : optional row gather on A (A row m -> A[aidx[m]])
//   act   : 0 none, 1 relu, 2 tanh, 3 sigmoid;  accum : add into existing C
// One wave per 32x32 output tile (2x2 WMMA tiles, A/B fragments reused).
// Rows/cols are CLAMPED (never guarded) so all loads are unconditional and
// in-bounds; bad rows/cols only pollute store-guarded outputs. Only the K
// tail uses value selects (cndmask, no EXEC divergence).
// ---------------------------------------------------------------------------
template <bool ALIGNED>
__global__ __launch_bounds__(128) void k_gemm(
    const float* __restrict__ A, int lda, const int* __restrict__ aidx,
    const float* __restrict__ W, int ldw, const float* __restrict__ bias,
    float* __restrict__ C, int ldc,
    int M, int N, int K, float alpha, int act, int accum)
{
  const int lane = threadIdx.x & 31;
  const int wave = threadIdx.x >> 5;
  const int ntn  = (N + 31) >> 5;
  const int ntm  = (M + 31) >> 5;
  const int tile = blockIdx.x * 4 + wave;
  if (tile >= ntm * ntn) return;            // wave-uniform exit: EXEC stays full
  const int tm0 = (tile / ntn) << 5;
  const int tn0 = (tile % ntn) << 5;

  const int mloc = lane & 15;
  const int ksel = lane >> 4;

  int row0 = tm0 + mloc;       if (row0 > M - 1) row0 = M - 1;
  int row1 = tm0 + 16 + mloc;  if (row1 > M - 1) row1 = M - 1;
  const long r0 = aidx ? (long)aidx[row0] : (long)row0;
  const long r1 = aidx ? (long)aidx[row1] : (long)row1;
  const int col0 = tn0 + mloc;
  const int col1 = tn0 + 16 + mloc;
  const int cc0 = (col0 > N - 1) ? N - 1 : col0;
  const int cc1 = (col1 > N - 1) ? N - 1 : col1;

  const float* __restrict__ pa0 = A + r0 * lda;
  const float* __restrict__ pa1 = A + r1 * lda;
  const float* __restrict__ pw0 = W + (long)cc0 * ldw;
  const float* __restrict__ pw1 = W + (long)cc1 * ldw;

  v8f c00 = {}, c01 = {}, c10 = {}, c11 = {};
  int kk = 0;
  for (; kk + 32 <= K; kk += 32) {
    v16bf a0, a1, b0, b1;
    if (ALIGNED) {
      a0 = a_frag_vec(pa0, kk, ksel);
      a1 = a_frag_vec(pa1, kk, ksel);
      b0 = b_frag_vec(pw0, kk, ksel);
      b1 = b_frag_vec(pw1, kk, ksel);
    } else {
#pragma unroll
      for (int i = 0; i < 16; ++i) {
        const int ka = kk + ((i >> 3) << 4) + (ksel << 3) + (i & 7);
        a0[i] = (__bf16)pa0[ka];
        a1[i] = (__bf16)pa1[ka];
        const int kb = kk + ((i >> 3) << 4) + (ksel << 3) + (i & 7);
        b0[i] = (__bf16)pw0[kb];
        b1[i] = (__bf16)pw1[kb];
      }
    }
    if (kk + 32 < K) {                      // uniform guard: stays in-bounds
      __builtin_prefetch(pa0 + kk + 32, 0, 3);   // gfx1250 global_prefetch_b8
      __builtin_prefetch(pw0 + kk + 32, 0, 3);
    }
    c00 = __builtin_amdgcn_wmma_f32_16x16x32_bf16(false, a0, false, b0, (short)0, c00, false, false);
    c01 = __builtin_amdgcn_wmma_f32_16x16x32_bf16(false, a0, false, b1, (short)0, c01, false, false);
    c10 = __builtin_amdgcn_wmma_f32_16x16x32_bf16(false, a1, false, b0, (short)0, c10, false, false);
    c11 = __builtin_amdgcn_wmma_f32_16x16x32_bf16(false, a1, false, b1, (short)0, c11, false, false);
  }
  if (kk < K) {                             // ragged-K tail: clamp + select
    v16bf a0, a1, b0, b1;
#pragma unroll
    for (int i = 0; i < 16; ++i) {
      const int ka  = kk + ((i >> 3) << 4) + (ksel << 3) + (i & 7);
      const int kac = ka < K ? ka : K - 1;
      const float av0 = pa0[kac], av1 = pa1[kac];
      a0[i] = (__bf16)(ka < K ? av0 : 0.f);
      a1[i] = (__bf16)(ka < K ? av1 : 0.f);
      const int kb  = kk + ((i >> 3) << 4) + (ksel << 3) + (i & 7);
      const int kbc = kb < K ? kb : K - 1;
      const float bv0 = pw0[kbc], bv1 = pw1[kbc];
      b0[i] = (__bf16)(kb < K ? bv0 : 0.f);
      b1[i] = (__bf16)(kb < K ? bv1 : 0.f);
    }
    c00 = __builtin_amdgcn_wmma_f32_16x16x32_bf16(false, a0, false, b0, (short)0, c00, false, false);
    c01 = __builtin_amdgcn_wmma_f32_16x16x32_bf16(false, a0, false, b1, (short)0, c01, false, false);
    c10 = __builtin_amdgcn_wmma_f32_16x16x32_bf16(false, a1, false, b0, (short)0, c10, false, false);
    c11 = __builtin_amdgcn_wmma_f32_16x16x32_bf16(false, a1, false, b1, (short)0, c11, false, false);
  }

  // D layout: VGPR r -> m = r + 8*(lane>=16), n = lane&15
  const float bb0 = (bias && col0 < N) ? bias[col0] : 0.f;
  const float bb1 = (bias && col1 < N) ? bias[col1] : 0.f;
#pragma unroll
  for (int t = 0; t < 4; ++t) {
    const v8f  c   = (t == 0) ? c00 : (t == 1) ? c01 : (t == 2) ? c10 : c11;
    const int  mb  = tm0 + ((t >> 1) << 4);
    const int  cx  = (t & 1) ? col1 : col0;
    const bool cok = cx < N;
    const float bb = (t & 1) ? bb1 : bb0;
#pragma unroll
    for (int r = 0; r < 8; ++r) {
      const int m = mb + r + (ksel << 3);
      if (m < M && cok) {
        float v = alpha * c[r] + bb;
        const long off = (long)m * ldc + cx;
        if (accum) v += C[off];
        if      (act == 1) v = v > 0.f ? v : 0.f;
        else if (act == 2) v = tanhf(v);
        else if (act == 3) v = 1.f / (1.f + __expf(-v));
        C[off] = v;
      }
    }
  }
}

// LayerNorm over HID=256, one block per row. Optional residual R, optional relu.
__global__ __launch_bounds__(256) void k_ln(
    const float* __restrict__ X, const float* __restrict__ R,
    const float* __restrict__ g, const float* __restrict__ b,
    float* __restrict__ Out, int relu)
{
  __shared__ float red[256];
  const int row = blockIdx.x, d = threadIdx.x;
  float v = X[(long)row * HID + d];
  if (R) v += R[(long)row * HID + d];
  red[d] = v; __syncthreads();
  for (int s = 128; s > 0; s >>= 1) { if (d < s) red[d] += red[d + s]; __syncthreads(); }
  const float mean = red[0] * (1.f / HID); __syncthreads();
  const float dv = v - mean;
  red[d] = dv * dv; __syncthreads();
  for (int s = 128; s > 0; s >>= 1) { if (d < s) red[d] += red[d + s]; __syncthreads(); }
  const float var = red[0] * (1.f / HID);
  float o = dv * rsqrtf(var + 1e-5f) * g[d] + b[d];
  if (relu) o = o > 0.f ? o : 0.f;
  Out[(long)row * HID + d] = o;
}

// Row softmax (in place), one block per row, arbitrary col count.
__global__ __launch_bounds__(256) void k_softmax_row(float* __restrict__ S, int cols)
{
  __shared__ float red[256];
  const long base = (long)blockIdx.x * cols;
  const int t = threadIdx.x;
  float mx = -3.0e38f;
  for (int c = t; c < cols; c += 256) mx = fmaxf(mx, S[base + c]);
  red[t] = mx; __syncthreads();
  for (int s = 128; s > 0; s >>= 1) { if (t < s) red[t] = fmaxf(red[t], red[t + s]); __syncthreads(); }
  mx = red[0]; __syncthreads();
  float sum = 0.f;
  for (int c = t; c < cols; c += 256) { float e = __expf(S[base + c] - mx); S[base + c] = e; sum += e; }
  red[t] = sum; __syncthreads();
  for (int s = 128; s > 0; s >>= 1) { if (t < s) red[t] += red[t + s]; __syncthreads(); }
  const float inv = 1.f / red[0];
  for (int c = t; c < cols; c += 256) S[base + c] *= inv;
}

__global__ void k_transpose(const float* __restrict__ in, float* __restrict__ out,
                            int rows, int cols)
{
  const long i = (long)blockIdx.x * blockDim.x + threadIdx.x;
  if (i < (long)rows * cols) {
    const int r = (int)(i / cols), c = (int)(i % cols);
    out[(long)c * rows + r] = in[i];
  }
}

__global__ void k_copy(const float* __restrict__ in, float* __restrict__ out, long n)
{
  const long i = (long)blockIdx.x * blockDim.x + threadIdx.x;
  if (i < n) out[i] = in[i];
}

// gcrow[d] = sum_n x[n][d] * w[n]
__global__ __launch_bounds__(256) void k_weighted_sum(
    const float* __restrict__ x, const float* __restrict__ w,
    float* __restrict__ out, int n)
{
  const int d = threadIdx.x;
  float acc = 0.f;
  for (int i = 0; i < n; ++i) acc += x[(long)i * HID + d] * w[i];
  out[d] = acc;
}

// colvec[d] = gate.b[d] + sum_j gate.w[d][256+j] * gcvec[j]
__global__ __launch_bounds__(256) void k_gate_colvec(
    const float* __restrict__ Wg, const float* __restrict__ bg,
    const float* __restrict__ gcvec, float* __restrict__ colvec)
{
  const int d = threadIdx.x;
  float acc = bg[d];
  for (int j = 0; j < HID; ++j) acc += Wg[(long)d * 512 + HID + j] * gcvec[j];
  colvec[d] = acc;
}

// comb = x*(1-g) + gc*g  (gc is a broadcast row)
__global__ void k_fuse(const float* __restrict__ x, const float* __restrict__ g,
                       const float* __restrict__ gcvec, float* __restrict__ comb)
{
  const long i = (long)blockIdx.x * blockDim.x + threadIdx.x;
  if (i < (long)NATOM * HID) {
    const int d = (int)(i & (HID - 1));
    const float gg = g[i];
    comb[i] = x[i] * (1.f - gg) + gcvec[d] * gg;
  }
}

// nb[m][k] = relu(nb[m][k] + xa[m/12][k])   (m in [0, 4096*12))
__global__ void k_addrelu12(float* __restrict__ nb, const float* __restrict__ xa)
{
  const long i = (long)blockIdx.x * blockDim.x + threadIdx.x;
  if (i < (long)NATOM * MAXNB * HID) {
    const long m = i >> 8;
    const int  k = (int)(i & (HID - 1));
    float v = nb[i] + xa[(m / MAXNB) * HID + k];
    nb[i] = v > 0.f ? v : 0.f;
  }
}

// softmax over 12 neighbor slots per (atom, head), then mean over 8 heads.
__global__ __launch_bounds__(32) void k_nbr_softmax(
    const float* __restrict__ sc, float* __restrict__ nw)
{
  __shared__ float sm[MAXNB * HEADS];
  const int n = blockIdx.x, t = threadIdx.x;
  if (t < HEADS) {
    float e[MAXNB]; float mx = -3.0e38f;
    for (int s = 0; s < MAXNB; ++s) {
      e[s] = sc[((long)n * MAXNB + s) * HEADS + t];
      mx = fmaxf(mx, e[s]);
    }
    float sum = 0.f;
    for (int s = 0; s < MAXNB; ++s) { e[s] = __expf(e[s] - mx); sum += e[s]; }
    const float inv = 1.f / sum;
    for (int s = 0; s < MAXNB; ++s) sm[s * HEADS + t] = e[s] * inv;
  }
  __syncthreads();
  if (t < MAXNB) {
    float a = 0.f;
    for (int h = 0; h < HEADS; ++h) a += sm[t * HEADS + h];
    nw[(long)n * MAXNB + t] = a * (1.f / HEADS);
  }
}

// z[n][d] = y[n][d] + sum_s x[a2b[n,s]][d] * nw[n,s] * edge[a2b[n,s]][d]
__global__ void k_msg(const float* __restrict__ y, const float* __restrict__ xin,
                      const float* __restrict__ edge, const int* __restrict__ a2b,
                      const float* __restrict__ nw, float* __restrict__ z)
{
  const long i = (long)blockIdx.x * blockDim.x + threadIdx.x;
  if (i < (long)NATOM * HID) {
    const int n = (int)(i >> 8), d = (int)(i & (HID - 1));
    float acc = y[i];
    for (int s = 0; s < MAXNB; ++s) {
      const int  idx = a2b[n * MAXNB + s];
      const float w = nw[n * MAXNB + s];
      acc += xin[(long)idx * HID + d] * w * edge[(long)idx * HID + d];
    }
    z[i] = acc;
  }
}

// edge[e][d] += x[b2a[e]][d]
__global__ void k_edge_update(float* __restrict__ edge, const float* __restrict__ x,
                              const int* __restrict__ b2a)
{
  const long i = (long)blockIdx.x * blockDim.x + threadIdx.x;
  if (i < (long)NBOND * HID) {
    const int e = (int)(i >> 8), d = (int)(i & (HID - 1));
    edge[i] += x[(long)b2a[e] * HID + d];
  }
}

// ---------------------------------------------------------------------------
static inline void gemm(hipStream_t st, const float* A, int lda, const int* aidx,
                        const float* W, int ldw, const float* bias,
                        float* C, int ldc, int M, int N, int K,
                        float alpha = 1.f, int act = 0, int accum = 0)
{
  const int tiles  = ((M + 31) / 32) * ((N + 31) / 32);
  const int blocks = (tiles + 3) / 4;
  const bool aligned = ((lda & 3) == 0) && ((ldw & 3) == 0);
  if (aligned)
    k_gemm<true><<<blocks, 128, 0, st>>>(A, lda, aidx, W, ldw, bias, C, ldc,
                                         M, N, K, alpha, act, accum);
  else
    k_gemm<false><<<blocks, 128, 0, st>>>(A, lda, aidx, W, ldw, bias, C, ldc,
                                          M, N, K, alpha, act, accum);
}
static inline int eb(long n) { return (int)((n + 255) / 256); }

extern "C" void kernel_launch(void* const* d_in, const int* in_sizes, int n_in,
                              void* d_out, int out_size, void* d_ws, size_t ws_size,
                              hipStream_t stream)
{
  (void)in_sizes; (void)n_in; (void)out_size; (void)ws_size;
  auto F = [&](int i) { return (const float*)d_in[i]; };
  const int* a2b = (const int*)d_in[0];
  const int* b2a = (const int*)d_in[1];
  const float* f_atoms = F(2);
  const float* f_bonds = F(3);

  // workspace layout (floats)
  float* ws    = (float*)d_ws;
  float* x     = ws;                    // [4096,256] current atom state
  float* x0    = x    + 1048576;        // [4096,256]
  float* edge  = x0   + 1048576;        // [8192,256]
  float* comb  = edge + 2097152;        // [4096,256]
  float* q     = comb + 1048576;        // [4096,256]  (reused as y post-MHA)
  float* kb    = q    + 1048576;        // [4096,256]
  float* vb    = kb   + 1048576;        // [4096,256]
  float* vT    = vb   + 1048576;        // [256,4096]
  float* attno = vT   + 1048576;        // [4096,256]
  float* tA    = attno+ 1048576;        // [4096,256] temp
  float* tB    = tA   + 1048576;        // [4096,256] temp (z)
  float* sm    = tB   + 1048576;        // small: 8192 floats
  float* wgc   = sm;                    // [4096]
  float* gcrow = sm + 4096;             // [256]
  float* gc2   = sm + 4352;             // [256]
  float* gcvec = sm + 4608;             // [256]
  float* colv  = sm + 4864;             // [256]
  float* nsc   = sm + 8192;             // [49152,8]
  float* nw    = nsc + 393216;          // [49152]
  float* big   = nw  + 49152;           // 16,777,216 f: scores / nb / ff1

  // ---- encoders: linear -> layernorm -> relu --------------------------------
  gemm(stream, f_atoms, 133, nullptr, F(5), 133, F(4), x, HID, NATOM, HID, 133);
  k_ln<<<NATOM, 256, 0, stream>>>(x, nullptr, F(7), F(6), x, 1);
  k_copy<<<eb(1048576), 256, 0, stream>>>(x, x0, 1048576);
  gemm(stream, f_bonds, 147, nullptr, F(9), 147, F(8), edge, HID, NBOND, HID, 147);
  k_ln<<<NBOND, 256, 0, stream>>>(edge, nullptr, F(11), F(10), edge, 1);

  const float iscale = 0.17677669529663687f;  // 1/sqrt(DH)

  for (int l = 0; l < 3; ++l) {
    const int P = 22 + l * 20;
    // P+0 ff1.b  P+1 ff1.w  P+2 ff2.b  P+3 ff2.w  P+4 k.b  P+5 k.w
    // P+6 n1b    P+7 n1g    P+8 n2b    P+9 n2g    P+10 na1.b P+11 na1.w
    // P+12 na2.b P+13 na2.w P+14 o.b   P+15 o.w   P+16 q.b  P+17 q.w
    // P+18 v.b   P+19 v.w

    // ---- global context ----------------------------------------------------
    gemm(stream, x, HID, nullptr, F(13), HID, F(12), tA, HID, NATOM, HID, HID, 1.f, 2);
    gemm(stream, tA, HID, nullptr, F(15), HID, F(14), wgc, 1, NATOM, 1, HID);
    k_softmax_row<<<1, 256, 0, stream>>>(wgc, NATOM);
    k_weighted_sum<<<1, 256, 0, stream>>>(x, wgc, gcrow, NATOM);
    gemm(stream, gcrow, HID, nullptr, F(17), HID, F(16), gc2, HID, 1, HID, HID, 1.f, 1);
    gemm(stream, gc2, HID, nullptr, F(19), HID, F(18), gcvec, HID, 1, HID, HID);

    // ---- gated fusion: g = sigmoid(x@Wl^T + (b + Wr@gc)) -------------------
    k_gate_colvec<<<1, 256, 0, stream>>>(F(21), F(20), gcvec, colv);
    gemm(stream, x, HID, nullptr, F(21), 512, colv, tB, HID, NATOM, HID, HID, 1.f, 3);
    k_fuse<<<eb(1048576), 256, 0, stream>>>(x, tB, gcvec, comb);

    // ---- MHA over comb -----------------------------------------------------
    gemm(stream, comb, HID, nullptr, F(P + 17), HID, F(P + 16), q,  HID, NATOM, HID, HID);
    gemm(stream, comb, HID, nullptr, F(P + 5),  HID, F(P + 4),  kb, HID, NATOM, HID, HID);
    gemm(stream, comb, HID, nullptr, F(P + 19), HID, F(P + 18), vb, HID, NATOM, HID, HID);
    k_transpose<<<eb(1048576), 256, 0, stream>>>(vb, vT, NATOM, HID);
    for (int h = 0; h < HEADS; ++h) {
      // S_h = (Q_h @ K_h^T) / sqrt(DH)  -> big [4096,4096]
      gemm(stream, q + h * DH, HID, nullptr, kb + h * DH, HID, nullptr,
           big, NATOM, NATOM, NATOM, DH, iscale);
      k_softmax_row<<<NATOM, 256, 0, stream>>>(big, NATOM);
      // O_h = P @ V_h  (W[n][k] = vT[h*DH+n][k] = V[k][h*DH+n])
      gemm(stream, big, NATOM, nullptr, vT + (long)h * DH * NATOM, NATOM, nullptr,
           attno + h * DH, HID, NATOM, DH, NATOM);
    }
    gemm(stream, attno, HID, nullptr, F(P + 15), HID, F(P + 14), tA, HID, NATOM, HID, HID);
    k_ln<<<NATOM, 256, 0, stream>>>(comb, tA, F(P + 7), F(P + 6), q, 0);  // y -> q

    // ---- neighbor attention -------------------------------------------------
    // xa = y @ na1.w[:, :256]^T + b ; nb = x[a2b] @ na1.w[:, 256:]^T
    gemm(stream, q, HID, nullptr, F(P + 11), 512, F(P + 10), tA, HID, NATOM, HID, HID);
    gemm(stream, x, HID, a2b, F(P + 11) + HID, 512, nullptr, big, HID,
         NATOM * MAXNB, HID, HID);
    k_addrelu12<<<eb((long)NATOM * MAXNB * HID), 256, 0, stream>>>(big, tA);
    gemm(stream, big, HID, nullptr, F(P + 13), HID, F(P + 12), nsc, HEADS,
         NATOM * MAXNB, HEADS, HID);
    k_nbr_softmax<<<NATOM, 32, 0, stream>>>(nsc, nw);
    k_msg<<<eb(1048576), 256, 0, stream>>>(q, x, edge, a2b, nw, tB);  // z -> tB

    // ---- FFN + LN ----------------------------------------------------------
    gemm(stream, tB, HID, nullptr, F(P + 1), HID, F(P + 0), big, 1024,
         NATOM, 1024, HID, 1.f, 1);
    gemm(stream, big, 1024, nullptr, F(P + 3), 1024, F(P + 2), tA, HID,
         NATOM, HID, 1024);
    k_ln<<<NATOM, 256, 0, stream>>>(tB, tA, F(P + 9), F(P + 8), x, 0);

    // ---- edge update -------------------------------------------------------
    k_edge_update<<<eb((long)NBOND * HID), 256, 0, stream>>>(edge, x, b2a);
  }

  // ---- output head: l1(concat(x,x0)) -> LN -> relu -> l2 -------------------
  gemm(stream, x,  HID, nullptr, F(83),       512, F(82), tA, HID, NATOM, HID, HID);
  gemm(stream, x0, HID, nullptr, F(83) + HID, 512, nullptr, tA, HID, NATOM, HID, HID,
       1.f, 0, 1);
  k_ln<<<NATOM, 256, 0, stream>>>(tA, nullptr, F(87), F(86), tA, 1);
  gemm(stream, tA, HID, nullptr, F(85), HID, F(84), (float*)d_out, HID,
       NATOM, HID, HID);
}
